// CrystalGraphNeuralNetwork_15865609191626
// MI455X (gfx1250) — matrix operand order repaired and checked by reading, not compiled, not run on hardware
//
#include <hip/hip_runtime.h>
#include <hip/hip_bf16.h>

typedef __attribute__((ext_vector_type(16))) _Float16 v16h;
typedef __attribute__((ext_vector_type(8)))  _Float16 v8h;
typedef __attribute__((ext_vector_type(8)))  float    v8f;
typedef __attribute__((ext_vector_type(4)))  int      v4i;

#define D_FEAT 128
#define AS1 __attribute__((address_space(1)))
#define AS3 __attribute__((address_space(3)))

// ---------------------------------------------------------------------------
// Pack a row-major f32 [N,128] activation matrix into the WMMA 16-bit A-matrix
// lane layout (ISA 7.12.2), f16:
//   packed[((rt*4 + kbi)*32 + lane)*16 + i] = in[row*128 + col]
//   row = rt*16 + (lane&15), half = lane>>4
//   col = kbi*32 + (i<8 ? half*8 + i : 16 + half*8 + (i-8))
// Consecutive threads write consecutive halves -> fully coalesced stores.
// ---------------------------------------------------------------------------
__global__ void cgnn_pack_a_kernel(const float* __restrict__ in,
                                   _Float16* __restrict__ out, int total) {
  int p = blockIdx.x * blockDim.x + threadIdx.x;
  const int stride = gridDim.x * blockDim.x;
  for (; p < total; p += stride) {
    const int i    = p & 15;
    const int lane = (p >> 4) & 31;
    const int kbi  = (p >> 9) & 3;
    const int rt   = p >> 11;
    const int half = lane >> 4;
    const int row  = (rt << 4) + (lane & 15);
    const int cib  = (i < 8) ? (half * 8 + i) : (16 + half * 8 + (i - 8));
    const int col  = (kbi << 5) + cib;
    out[p] = (_Float16)in[(size_t)row * D_FEAT + col];
  }
}

// ---------------------------------------------------------------------------
// Pack a row-major f32 [128,128] weight matrix into the WMMA 16-bit B-matrix
// lane layout, f16:
//   packed[((ct*4 + kbi)*32 + lane)*16 + i] = in[k*128 + n]
//   k = kbi*32 + (i<8 ? (lane>>4)*8 + i : 16 + (lane>>4)*8 + (i-8)),  n = ct*16 + (lane&15)
// ---------------------------------------------------------------------------
__global__ void cgnn_pack_w_kernel(const float* __restrict__ in,
                                   _Float16* __restrict__ out, int total) {
  int p = blockIdx.x * blockDim.x + threadIdx.x;
  const int stride = gridDim.x * blockDim.x;
  for (; p < total; p += stride) {
    const int i    = p & 15;
    const int lane = (p >> 4) & 31;
    const int kbi  = (p >> 9) & 3;
    const int ct   = p >> 11;
    const int k    = (kbi << 5) + ((i < 8) ? (((lane >> 4) << 3) + i) : (16 + ((lane >> 4) << 3) + (i - 8)));
    const int n    = (ct << 4) + (lane & 15);
    out[p] = (_Float16)in[(size_t)k * D_FEAT + n];
  }
}

// ---------------------------------------------------------------------------
// zero-fill f32 (grid-stride)
// ---------------------------------------------------------------------------
__global__ void cgnn_clear_kernel(float* __restrict__ p, int n) {
  int i = blockIdx.x * blockDim.x + threadIdx.x;
  const int stride = gridDim.x * blockDim.x;
  for (; i < n; i += stride) p[i] = 0.0f;
}

// ---------------------------------------------------------------------------
// xt[N,128] = A @ B with pre-packed f16 operands, f32 accumulate via WMMA.
// One wave per 16x16 tile, 8 waves/block share one 4KB packed A row-tile which
// is staged into LDS via global_load_async_to_lds_b128 (ASYNCcnt path), with
// a plain b128+ds_store fallback if the builtin is absent.
// Inner loop: 1x ds b128x2 (A) + 1x global b128x2 (B) + 1x WMMA, x4 K-steps.
// ---------------------------------------------------------------------------
__global__ void cgnn_gemm_wmma_kernel(const _Float16* __restrict__ Apack,
                                      const _Float16* __restrict__ Wpack,
                                      float* __restrict__ C, int N) {
  __shared__ _Float16 As[16 * D_FEAT]; // 4KB packed A tile
  const int rt = blockIdx.x;
  if (rt * 16 >= N) return;
  const int t = threadIdx.x;
  const _Float16* gA = Apack + (size_t)rt * (16 * D_FEAT);

#if __has_builtin(__builtin_amdgcn_global_load_async_to_lds_b128)
  __builtin_amdgcn_global_load_async_to_lds_b128(
      (AS1 v4i*)(gA + t * 8), (AS3 v4i*)(As + t * 8), 0, 0);
#if __has_builtin(__builtin_amdgcn_s_wait_asynccnt)
  __builtin_amdgcn_s_wait_asynccnt(0);
#else
  asm volatile("s_wait_asynccnt 0" ::: "memory");
#endif
#else
  *(v8h*)(As + t * 8) = *(const v8h*)(gA + t * 8);
#endif
  __syncthreads();

  const int lane = t & 31;
  const int wave = t >> 5; // column tile 0..7
  const _Float16* wbase = Wpack + ((size_t)wave * 4 * 32) * 16;

  v8f c = {};
#pragma unroll
  for (int kbi = 0; kbi < 4; ++kbi) {
    const v16h a = *(const v16h*)(As + (kbi * 32 + lane) * 16);
    const v16h b = *(const v16h*)(wbase + (size_t)(kbi * 32 + lane) * 16);
    c = __builtin_amdgcn_wmma_f32_16x16x32_f16(false, a, false, b,
                                               (short)0, c, false, false);
  }

  const int half = lane >> 4;
  const int l15  = lane & 15;
#pragma unroll
  for (int r = 0; r < 8; ++r)
    C[(size_t)(rt * 16 + r + 8 * half) * D_FEAT + wave * 16 + l15] = c[r];
}

// ---------------------------------------------------------------------------
// Edge scatter: one wave per edge (grid-stride), lane handles 4 features.
// gate recomputed in registers: sigmoid(edge_attr[e]*eW[d] + eb[d]).
// acc[dst] += xt[src]*gate via global_atomic_add_f32 (xt/acc are L2-resident).
// Next iteration's xt row is prefetched (global_prefetch_b8) to hide the
// random-gather latency behind the current edge's VALU + atomic work.
// ---------------------------------------------------------------------------
__global__ void cgnn_edge_scatter_kernel(const float* __restrict__ xt,
                                         const int* __restrict__ src,
                                         const int* __restrict__ dst,
                                         const float* __restrict__ ea,
                                         const float* __restrict__ eW,
                                         const float* __restrict__ eb,
                                         float* __restrict__ acc, int E) {
  const int nWaves = (gridDim.x * blockDim.x) >> 5;
  const int lane = threadIdx.x & 31;
  const int f = lane * 4;
  const float4 wv = *(const float4*)(eW + f);
  const float4 bv = *(const float4*)(eb + f);

  for (int e = (blockIdx.x * blockDim.x + threadIdx.x) >> 5; e < E; e += nWaves) {
    const int en = e + nWaves;
    if (en < E)
      __builtin_prefetch(xt + (size_t)src[en] * D_FEAT + f, 0, 3);

    const int s = src[e];
    const int d = dst[e];
    const float a = ea[e];

    const float4 xv = *(const float4*)(xt + (size_t)s * D_FEAT + f);
    const float g0 = 1.0f / (1.0f + __expf(-(a * wv.x + bv.x)));
    const float g1 = 1.0f / (1.0f + __expf(-(a * wv.y + bv.y)));
    const float g2 = 1.0f / (1.0f + __expf(-(a * wv.z + bv.z)));
    const float g3 = 1.0f / (1.0f + __expf(-(a * wv.w + bv.w)));

    float* out = acc + (size_t)d * D_FEAT + f;
    atomicAdd(out + 0, xv.x * g0);
    atomicAdd(out + 1, xv.y * g1);
    atomicAdd(out + 2, xv.z * g2);
    atomicAdd(out + 3, xv.w * g3);
  }
}

// ---------------------------------------------------------------------------
// Post (layers 0,1): h = relu(acc + bias) written straight into the packed
// f16 A-layout for the next layer's GEMM. Iterates packed index -> coalesced
// f16 stores; acc reads are gathers within an 8KB row-tile window (L2 hits).
// ---------------------------------------------------------------------------
__global__ void cgnn_post_pack_kernel(const float* __restrict__ acc,
                                      const float* __restrict__ bias,
                                      _Float16* __restrict__ out, int total) {
  int p = blockIdx.x * blockDim.x + threadIdx.x;
  const int stride = gridDim.x * blockDim.x;
  for (; p < total; p += stride) {
    const int i    = p & 15;
    const int lane = (p >> 4) & 31;
    const int kbi  = (p >> 9) & 3;
    const int rt   = p >> 11;
    const int half = lane >> 4;
    const int row  = (rt << 4) + (lane & 15);
    const int cib  = (i < 8) ? (half * 8 + i) : (16 + half * 8 + (i - 8));
    const int col  = (kbi << 5) + cib;
    float v = acc[(size_t)row * D_FEAT + col] + bias[col];
    v = v > 0.0f ? v : 0.0f;
    out[p] = (_Float16)v;
  }
}

// ---------------------------------------------------------------------------
// Mean (layer 2 only): gmean[d] += sum_rows relu(acc[r,d]+bias[d]) / N.
// Per-thread column partial sums -> one atomic per thread.
// ---------------------------------------------------------------------------
__global__ void cgnn_mean_kernel(const float* __restrict__ acc,
                                 const float* __restrict__ bias,
                                 float* __restrict__ gmean, int N, float invN) {
  const int t = blockIdx.x * blockDim.x + threadIdx.x;
  const int d = t & (D_FEAT - 1);
  const int rowStep = (gridDim.x * blockDim.x) >> 7;
  const float b = bias[d];
  float sum = 0.0f;
  for (int r = t >> 7; r < N; r += rowStep) {
    float v = acc[(size_t)r * D_FEAT + d] + b;
    sum += (v > 0.0f ? v : 0.0f);
  }
  atomicAdd(&gmean[d], sum * invN);
}

// ---------------------------------------------------------------------------
// Final: out = dot(gmean, outW) + outb  (128-thread LDS tree reduction)
// ---------------------------------------------------------------------------
__global__ void cgnn_final_kernel(const float* __restrict__ gmean,
                                  const float* __restrict__ outW,
                                  const float* __restrict__ outb,
                                  float* __restrict__ out) {
  __shared__ float red[D_FEAT];
  const int t = threadIdx.x;
  red[t] = gmean[t] * outW[t];
  __syncthreads();
  for (int s = D_FEAT / 2; s > 0; s >>= 1) {
    if (t < s) red[t] += red[t + s];
    __syncthreads();
  }
  if (t == 0) out[0] = red[0] + outb[0];
}

// ---------------------------------------------------------------------------
extern "C" void kernel_launch(void* const* d_in, const int* in_sizes, int n_in,
                              void* d_out, int out_size, void* d_ws, size_t ws_size,
                              hipStream_t stream) {
  const float* x     = (const float*)d_in[0];
  const int*   eidx  = (const int*)d_in[1];   // [2, E] int32
  const float* eattr = (const float*)d_in[2]; // [E, 1]
  const float* W[3]  = {(const float*)d_in[3],  (const float*)d_in[7],  (const float*)d_in[11]};
  const float* bl[3] = {(const float*)d_in[4],  (const float*)d_in[8],  (const float*)d_in[12]};
  const float* eW[3] = {(const float*)d_in[5],  (const float*)d_in[9],  (const float*)d_in[13]};
  const float* ebv[3]= {(const float*)d_in[6],  (const float*)d_in[10], (const float*)d_in[14]};
  const float* outW  = (const float*)d_in[15];
  const float* outb  = (const float*)d_in[16];
  float* out = (float*)d_out;

  const int N = in_sizes[0] / D_FEAT; // 50000
  const int E = in_sizes[2];          // 800000
  const int* src = eidx;
  const int* dst = eidx + E;

  // Workspace layout (~64.1 MB, all offsets 32B aligned)
  char* ws = (char*)d_ws;
  size_t o = 0;
  _Float16* xh = (_Float16*)(ws + o); o += (size_t)N * D_FEAT * sizeof(_Float16); // packed A
  float*    xt = (float*)(ws + o);    o += (size_t)N * D_FEAT * sizeof(float);
  float*    acc= (float*)(ws + o);    o += (size_t)N * D_FEAT * sizeof(float);
  _Float16* Wh = (_Float16*)(ws + o); o += (size_t)3 * D_FEAT * D_FEAT * sizeof(_Float16); // packed B
  float* gmean = (float*)(ws + o);    o += D_FEAT * sizeof(float);

  const int ND = N * D_FEAT;
  const int WD = D_FEAT * D_FEAT;

  // One-time per call: pack activations + weights, clear mean accumulator
  cgnn_pack_a_kernel<<<2048, 256, 0, stream>>>(x, xh, ND);
  for (int l = 0; l < 3; ++l)
    cgnn_pack_w_kernel<<<64, 256, 0, stream>>>(W[l], Wh + (size_t)l * WD, WD);
  cgnn_clear_kernel<<<1, 128, 0, stream>>>(gmean, D_FEAT);

  const int gemmBlocks = (N + 15) / 16;

  for (int l = 0; l < 3; ++l) {
    cgnn_clear_kernel<<<2048, 256, 0, stream>>>(acc, ND);
    cgnn_gemm_wmma_kernel<<<gemmBlocks, 256, 0, stream>>>(
        xh, Wh + (size_t)l * WD, xt, N);
    cgnn_edge_scatter_kernel<<<2048, 256, 0, stream>>>(
        xt, src, dst, eattr, eW[l], ebv[l], acc, E);
    if (l < 2) {
      cgnn_post_pack_kernel<<<2048, 256, 0, stream>>>(acc, bl[l], xh, ND);
    } else {
      cgnn_mean_kernel<<<512, 256, 0, stream>>>(acc, bl[l], gmean, N, 1.0f / (float)N);
    }
  }

  cgnn_final_kernel<<<1, D_FEAT, 0, stream>>>(gmean, outW, outb, out);
}
